// MSFEICL_68118181314817
// MI455X (gfx1250) — matrix operand order repaired and checked by reading, not compiled, not run on hardware
//
#include <hip/hip_runtime.h>

// ---------------- types for CDNA5 WMMA ----------------
typedef __bf16 bf16;
typedef __attribute__((ext_vector_type(16))) __bf16 v16bf;
typedef __attribute__((ext_vector_type(8)))  __bf16 bf16x8;
typedef __attribute__((ext_vector_type(8)))  float  v8f;

// ---------------- problem constants ----------------
constexpr int NN = 8000;      // nodes
constexpr int EE = 262144;    // edges
constexpr int FM = 256;       // embedding dim / feature-map height
constexpr int PJW = 1280;     // 5*EMB
// padded feature widths (multiple of 32 for WMMA K, multiple of 16 for N tiles)
constexpr int CTD_P = 320, KMER_P = 352, DOC_P = 256, ROLE_P = 128;

// =====================================================================
// bf16 WMMA GEMM:  C[M=8000, Nout] = A[M,Kp] * Bt[Nout,Kp]^T + bias
// A row-major bf16 (lda=Kp), Bt row-major bf16 [Nout,Kp] (B transposed),
// C fp32 with stride ldc and column offset coloff.
// Each wave owns a 64(M) x 16(N) strip: 4 accumulators sharing one B panel
// per k-step (B loaded once per 4 WMMAs -> 2.5 b128 loads per WMMA).
// Per-lane addressing matches CDNA5 16-bit WMMA VGPR striping:
//   A 16x32: lanes 0-15 row M=l, elems 0-7 = K0..7, 8-15 = K16..23;
//            lanes 16-31 same rows, K+8 / K+24.
//   B 32x16: lanes 0-15 col N=l, elems 0-7 = K0..7, 8-15 = K16..23; lanes 16-31 K+8 / K+24.
//   D 16x16: vgpr r -> M = r + 8*(lane>=16), N = lane&15.
// =====================================================================
__global__ void k_gemm_bf16(const bf16* __restrict__ A, const bf16* __restrict__ Bt,
                            const float* __restrict__ bias, float* __restrict__ C,
                            int K, int ldc, int coloff) {
  const int lane = threadIdx.x & 31;
  const int half = lane >> 4;
  const int l    = lane & 15;
  const int m0 = (blockIdx.x * blockDim.y + threadIdx.y) * 64;   // 64-row strip
  const int n0 = blockIdx.y * 16;

  const bf16* a0 = A + (size_t)(m0 +  0 + l) * K;
  const bf16* a1 = A + (size_t)(m0 + 16 + l) * K;
  const bf16* a2 = A + (size_t)(m0 + 32 + l) * K;
  const bf16* a3 = A + (size_t)(m0 + 48 + l) * K;
  const bf16* br = Bt + (size_t)(n0 + l) * K;

  const int alo = half * 8, ahi = 16 + half * 8;      // A elem offsets per lane-half
  const int blo = half * 8, bhi = 16 + half * 8;      // B elem offsets per lane-half

  v8f c0 = {}, c1 = {}, c2 = {}, c3 = {};
  for (int k0 = 0; k0 < K; k0 += 32) {
    union U { v16bf v; bf16x8 h[2]; } b, t0, t1, t2, t3;
    b.h[0]  = *(const bf16x8*)(br + k0 + blo);
    b.h[1]  = *(const bf16x8*)(br + k0 + bhi);
    t0.h[0] = *(const bf16x8*)(a0 + k0 + alo);
    t0.h[1] = *(const bf16x8*)(a0 + k0 + ahi);
    t1.h[0] = *(const bf16x8*)(a1 + k0 + alo);
    t1.h[1] = *(const bf16x8*)(a1 + k0 + ahi);
    t2.h[0] = *(const bf16x8*)(a2 + k0 + alo);
    t2.h[1] = *(const bf16x8*)(a2 + k0 + ahi);
    t3.h[0] = *(const bf16x8*)(a3 + k0 + alo);
    t3.h[1] = *(const bf16x8*)(a3 + k0 + ahi);
    // gfx1250 prefetch path (global_prefetch_b8), 8 k-steps ahead
    __builtin_prefetch(br + k0 + 256, 0, 0);
    __builtin_prefetch(a0 + k0 + 256, 0, 0);
    c0 = __builtin_amdgcn_wmma_f32_16x16x32_bf16(false, t0.v, false, b.v, (short)0, c0, false, false);
    c1 = __builtin_amdgcn_wmma_f32_16x16x32_bf16(false, t1.v, false, b.v, (short)0, c1, false, false);
    c2 = __builtin_amdgcn_wmma_f32_16x16x32_bf16(false, t2.v, false, b.v, (short)0, c2, false, false);
    c3 = __builtin_amdgcn_wmma_f32_16x16x32_bf16(false, t3.v, false, b.v, (short)0, c3, false, false);
  }
  const float bn = bias ? bias[n0 + l] : 0.f;
  float* cb = C + (size_t)coloff + (size_t)n0 + l;
#pragma unroll
  for (int r = 0; r < 8; ++r) {
    const int mr = m0 + r + half * 8;
    cb[(size_t)(mr +  0) * ldc] = c0[r] + bn;
    cb[(size_t)(mr + 16) * ldc] = c1[r] + bn;
    cb[(size_t)(mr + 32) * ldc] = c2[r] + bn;
    cb[(size_t)(mr + 48) * ldc] = c3[r] + bn;
  }
}

// ---------------- elementwise / prep kernels ----------------
__global__ void k_cast_pad(const float* __restrict__ src, bf16* __restrict__ dst,
                           int M, int K, int Kp, int relu) {
  const size_t i = (size_t)blockIdx.x * blockDim.x + threadIdx.x;
  if (i >= (size_t)M * Kp) return;
  const int m = (int)(i / Kp), k = (int)(i % Kp);
  float v = (k < K) ? src[(size_t)m * K + k] : 0.f;
  if (relu) v = fmaxf(v, 0.f);
  dst[i] = (bf16)v;
}

// src [K,N] row-major -> dst [Np,Kp] row-major (B transposed, zero-padded)
__global__ void k_tcast_pad(const float* __restrict__ src, bf16* __restrict__ dst,
                            int K, int N, int Kp, int Np) {
  const size_t i = (size_t)blockIdx.x * blockDim.x + threadIdx.x;
  if (i >= (size_t)Np * Kp) return;
  const int n = (int)(i / Kp), k = (int)(i % Kp);
  float v = (k < K && n < N) ? src[(size_t)k * N + n] : 0.f;
  dst[i] = (bf16)v;
}

__global__ void k_pad_bias(const float* __restrict__ b, float* __restrict__ bp, int N, int Np) {
  const int i = threadIdx.x;
  if (i < Np) bp[i] = (i < N) ? b[i] : 0.f;
}

// ---------------- GCN degree / aggregation ----------------
__global__ void k_fill1(float* d, int n) {
  const int i = blockIdx.x * blockDim.x + threadIdx.x;
  if (i < n) d[i] = 1.f;   // self-loop contributes 1 to every degree
}
__global__ void k_degacc(const int* __restrict__ dstI, float* deg, int e) {
  const int i = blockIdx.x * blockDim.x + threadIdx.x;
  if (i < e) unsafeAtomicAdd(&deg[dstI[i]], 1.f);
}
__global__ void k_dinv(float* d, int n) {
  const int i = blockIdx.x * blockDim.x + threadIdx.x;
  if (i < n) d[i] = rsqrtf(d[i]);   // deg >= 1 always (self loops)
}
// agg = h * dinv^2 (self loop) + bias
__global__ void k_agg_init(const float* __restrict__ h, const float* __restrict__ dinv,
                           const float* __restrict__ bias, float* __restrict__ agg, int Fp) {
  const size_t i = (size_t)blockIdx.x * blockDim.x + threadIdx.x;
  if (i >= (size_t)NN * Fp) return;
  const int n = (int)(i / Fp), f = (int)(i % Fp);
  agg[i] = h[i] * dinv[n] * dinv[n] + bias[f];
}
// agg[dst] += h[src] * dinv[src]*dinv[dst] (one block per edge)
__global__ void k_agg_edges(const int* __restrict__ srcI, const int* __restrict__ dstI,
                            const float* __restrict__ h, const float* __restrict__ dinv,
                            float* __restrict__ agg, int Fp) {
  const int e = blockIdx.x;
  const int s = srcI[e], d = dstI[e];
  const float norm = dinv[s] * dinv[d];
  const float* hs = h + (size_t)s * Fp;
  float* ad = agg + (size_t)d * Fp;
  for (int f = threadIdx.x; f < Fp; f += blockDim.x)
    unsafeAtomicAdd(&ad[f], hs[f] * norm);
}

// ---------------- CBAM ----------------
// per-channel partial sum/max over FM*NN elements; grid = (PB, 5)
__global__ void k_ca_partial(const float* __restrict__ pj,
                             float* __restrict__ psum, float* __restrict__ pmax) {
  __shared__ float ss[256], sm[256];
  const int c = blockIdx.y, tid = threadIdx.x;
  float s = 0.f, m = -3.402823466e38f;
  const size_t tot = (size_t)NN * FM;
  for (size_t i = (size_t)blockIdx.x * blockDim.x + tid; i < tot;
       i += (size_t)gridDim.x * blockDim.x) {
    const size_t n = i >> 8;               // [n][h] bijection, h = i & 255
    const float v = pj[n * PJW + c * 256 + (int)(i & 255)];
    s += v; m = fmaxf(m, v);
  }
  ss[tid] = s; sm[tid] = m;
  __syncthreads();
  for (int o = 128; o > 0; o >>= 1) {
    if (tid < o) { ss[tid] += ss[tid + o]; sm[tid] = fmaxf(sm[tid], sm[tid + o]); }
    __syncthreads();
  }
  if (tid == 0) { psum[c * gridDim.x + blockIdx.x] = ss[0];
                  pmax[c * gridDim.x + blockIdx.x] = sm[0]; }
}

__global__ void k_ca_finish(const float* __restrict__ psum, const float* __restrict__ pmax,
                            const float* __restrict__ w1, const float* __restrict__ w2,
                            float* __restrict__ ca, int PB) {
  if (threadIdx.x != 0 || blockIdx.x != 0) return;
  float avg[5], mx[5], t[5], oa[5], om[5];
  for (int c = 0; c < 5; ++c) {
    float s = 0.f, m = -3.402823466e38f;
    for (int b = 0; b < PB; ++b) { s += psum[c * PB + b]; m = fmaxf(m, pmax[c * PB + b]); }
    avg[c] = s / (float)((size_t)NN * FM); mx[c] = m;
  }
  for (int j = 0; j < 5; ++j) { float a = 0; for (int k = 0; k < 5; ++k) a += avg[k] * w1[k*5+j]; t[j] = fmaxf(a, 0.f); }
  for (int j = 0; j < 5; ++j) { float a = 0; for (int k = 0; k < 5; ++k) a += t[k] * w2[k*5+j]; oa[j] = a; }
  for (int j = 0; j < 5; ++j) { float a = 0; for (int k = 0; k < 5; ++k) a += mx[k] * w1[k*5+j]; t[j] = fmaxf(a, 0.f); }
  for (int j = 0; j < 5; ++j) { float a = 0; for (int k = 0; k < 5; ++k) a += t[k] * w2[k*5+j]; om[j] = a; }
  for (int c = 0; c < 5; ++c) ca[c] = 1.f / (1.f + __expf(-(oa[c] + om[c])));
}

// s[0] = mean_c(x*ca), s[1] = max_c(x*ca) over channels; layout [2][FM][NN]
__global__ void k_spool(const float* __restrict__ pj, const float* __restrict__ ca,
                        float* __restrict__ s) {
  const size_t i = (size_t)blockIdx.x * blockDim.x + threadIdx.x;
  if (i >= (size_t)FM * NN) return;
  const int h = (int)(i / NN), n = (int)(i % NN);
  float su = 0.f, m = -3.402823466e38f;
#pragma unroll
  for (int c = 0; c < 5; ++c) {
    const float v = pj[(size_t)n * PJW + c * 256 + h] * ca[c];
    su += v; m = fmaxf(m, v);
  }
  s[i] = su * 0.2f;
  s[(size_t)FM * NN + i] = m;
}

// 7x7 spatial conv (+sigmoid) fused with relu + 1x1 channel conv; writes x.T
__global__ void k_sconv(const float* __restrict__ s, const float* __restrict__ pj,
                        const float* __restrict__ ca, const float* __restrict__ saw,
                        const float* __restrict__ sab, const float* __restrict__ wcnn,
                        const float* __restrict__ bcnn, float* __restrict__ out) {
  const size_t i = (size_t)blockIdx.x * blockDim.x + threadIdx.x;
  if (i >= (size_t)FM * NN) return;
  const int h = (int)(i / NN), n = (int)(i % NN);
  float a = sab[0];
  for (int ch = 0; ch < 2; ++ch)
    for (int r = 0; r < 7; ++r) {
      const int hh = h - 3 + r;
      if (hh < 0 || hh >= FM) continue;
      const float* srow = s + (size_t)ch * FM * NN + (size_t)hh * NN;
      const float* wrow = saw + ch * 49 + r * 7;
      for (int q = 0; q < 7; ++q) {
        const int nn = n - 3 + q;
        if (nn < 0 || nn >= NN) continue;
        a += wrow[q] * srow[nn];
      }
    }
  const float sig = 1.f / (1.f + __expf(-a));
  float o = bcnn[0];
#pragma unroll
  for (int c = 0; c < 5; ++c)
    o += wcnn[c] * fmaxf(pj[(size_t)n * PJW + c * 256 + h] * ca[c] * sig, 0.f);
  out[(size_t)n * FM + h] = o;   // x.T: [N, FM]
}

// =====================================================================
extern "C" void kernel_launch(void* const* d_in, const int* in_sizes, int n_in,
                              void* d_out, int out_size, void* d_ws, size_t ws_size,
                              hipStream_t stream) {
  (void)in_sizes; (void)n_in; (void)out_size; (void)ws_size;
  const float* x_ctd      = (const float*)d_in[0];
  const float* x_doc      = (const float*)d_in[1];
  const float* x_kmer     = (const float*)d_in[2];
  const float* x_role     = (const float*)d_in[3];
  const float* sim        = (const float*)d_in[4];
  const float* w_gcn_ctd  = (const float*)d_in[5];   const float* b_gcn_ctd  = (const float*)d_in[6];
  const float* w_gcn_kmer = (const float*)d_in[7];   const float* b_gcn_kmer = (const float*)d_in[8];
  const float* w_gcn_doc  = (const float*)d_in[9];   const float* b_gcn_doc  = (const float*)d_in[10];
  const float* w_gcn_role = (const float*)d_in[11];  const float* b_gcn_role = (const float*)d_in[12];
  const float* w_p_ctd    = (const float*)d_in[13];  const float* b_p_ctd    = (const float*)d_in[14];
  const float* w_p_kmer   = (const float*)d_in[15];  const float* b_p_kmer   = (const float*)d_in[16];
  const float* w_p_doc    = (const float*)d_in[17];  const float* b_p_doc    = (const float*)d_in[18];
  const float* w_p_role   = (const float*)d_in[19];  const float* b_p_role   = (const float*)d_in[20];
  const float* w_pj_ctd   = (const float*)d_in[21];  const float* b_pj_ctd   = (const float*)d_in[22];
  const float* w_pj_doc   = (const float*)d_in[23];  const float* b_pj_doc   = (const float*)d_in[24];
  const float* w_pj_kmer  = (const float*)d_in[25];  const float* b_pj_kmer  = (const float*)d_in[26];
  const float* w_pj_role  = (const float*)d_in[27];  const float* b_pj_role  = (const float*)d_in[28];
  const float* w_pj_sim   = (const float*)d_in[29];  const float* b_pj_sim   = (const float*)d_in[30];
  const float* w_fc_ban   = (const float*)d_in[31];  const float* b_fc_ban   = (const float*)d_in[32];
  const float* w_cnn      = (const float*)d_in[33];  const float* b_cnn      = (const float*)d_in[34];
  const float* ca_w1      = (const float*)d_in[35];  const float* ca_w2      = (const float*)d_in[36];
  const float* sa_w       = (const float*)d_in[37];  const float* sa_b       = (const float*)d_in[38];
  const int*   edges      = (const int*)d_in[39];
  const int* srcI = edges;
  const int* dstI = edges + EE;

  float* out_xT  = (float*)d_out;                       // [8000,256]
  float* out_xm  = out_xT  + (size_t)NN * FM;           // [8000,1024]
  float* out_raw = out_xm  + (size_t)NN * 1024;         // [8000,256]

  // -------- carve workspace (256B aligned regions) --------
  char* p = (char*)d_ws;
  auto carve = [&](size_t bytes) -> char* {
    char* r = p; p += (bytes + 255) & ~(size_t)255; return r;
  };
  bf16* xb_ctd  = (bf16*)carve((size_t)NN * CTD_P  * 2);
  bf16* xb_doc  = (bf16*)carve((size_t)NN * DOC_P  * 2);
  bf16* xb_kmer = (bf16*)carve((size_t)NN * KMER_P * 2);
  bf16* xb_role = (bf16*)carve((size_t)NN * ROLE_P * 2);
  bf16* sim_bf  = (bf16*)carve((size_t)NN * NN * 2);          // 128MB, L2-resident
  bf16* wt_gcn_ctd  = (bf16*)carve((size_t)CTD_P  * CTD_P  * 2);
  bf16* wt_gcn_kmer = (bf16*)carve((size_t)KMER_P * KMER_P * 2);
  bf16* wt_gcn_doc  = (bf16*)carve((size_t)DOC_P  * DOC_P  * 2);
  bf16* wt_gcn_role = (bf16*)carve((size_t)ROLE_P * ROLE_P * 2);
  bf16* wt_p_ctd    = (bf16*)carve((size_t)256 * CTD_P  * 2);
  bf16* wt_p_kmer   = (bf16*)carve((size_t)256 * KMER_P * 2);
  bf16* wt_p_doc    = (bf16*)carve((size_t)256 * DOC_P  * 2);
  bf16* wt_p_role   = (bf16*)carve((size_t)256 * ROLE_P * 2);
  bf16* wt_pj_ctd   = (bf16*)carve((size_t)256 * CTD_P  * 2);
  bf16* wt_pj_doc   = (bf16*)carve((size_t)256 * DOC_P  * 2);
  bf16* wt_pj_kmer  = (bf16*)carve((size_t)256 * KMER_P * 2);
  bf16* wt_pj_role  = (bf16*)carve((size_t)256 * ROLE_P * 2);
  bf16* wt_pj_sim   = (bf16*)carve((size_t)256 * NN * 2);
  bf16* wt_fc       = (bf16*)carve((size_t)256 * PJW * 2);
  float* bp_ctd  = (float*)carve(CTD_P  * 4);
  float* bp_kmer = (float*)carve(KMER_P * 4);
  float* deg     = (float*)carve((size_t)NN * 4);             // becomes dinv
  float* hbuf    = (float*)carve((size_t)NN * KMER_P * 4);
  float* agg     = (float*)carve((size_t)NN * KMER_P * 4);
  bf16*  aggb    = (bf16*)carve((size_t)NN * KMER_P * 2);
  float* pj      = (float*)carve((size_t)NN * PJW * 4);
  bf16*  pj_bf   = (bf16*)carve((size_t)NN * PJW * 2);
  float* sbuf    = (float*)carve((size_t)2 * FM * NN * 4);
  float* psum    = (float*)carve(5 * 64 * 4);
  float* pmax    = (float*)carve(5 * 64 * 4);
  float* ca      = (float*)carve(5 * 4);

  const int T = 256;
  auto nb = [](size_t n, int t) { return (unsigned)((n + t - 1) / t); };
  auto gemm = [&](const bf16* A, const bf16* Bt, const float* bias, float* C,
                  int K, int Nout, int ldc, int coloff) {
    // 125 64-row wave strips = 25 blocks x 5 waves; grid.y = N tiles
    dim3 grid(25, Nout / 16), blk(32, 5);
    k_gemm_bf16<<<grid, blk, 0, stream>>>(A, Bt, bias, C, K, ldc, coloff);
  };

  // -------- phase A: casts / transposed weights / degree --------
  k_cast_pad<<<nb((size_t)NN*CTD_P, T), T, 0, stream>>>(x_ctd,  xb_ctd,  NN, 300, CTD_P, 0);
  k_cast_pad<<<nb((size_t)NN*DOC_P, T), T, 0, stream>>>(x_doc,  xb_doc,  NN, 256, DOC_P, 0);
  k_cast_pad<<<nb((size_t)NN*KMER_P,T), T, 0, stream>>>(x_kmer, xb_kmer, NN, 340, KMER_P,0);
  k_cast_pad<<<nb((size_t)NN*ROLE_P,T), T, 0, stream>>>(x_role, xb_role, NN, 128, ROLE_P,0);
  k_cast_pad<<<nb((size_t)NN*NN,    T), T, 0, stream>>>(sim,    sim_bf,  NN, NN,  NN,    0);

  k_tcast_pad<<<nb((size_t)CTD_P*CTD_P,  T), T, 0, stream>>>(w_gcn_ctd,  wt_gcn_ctd,  300, 300, CTD_P,  CTD_P);
  k_tcast_pad<<<nb((size_t)KMER_P*KMER_P,T), T, 0, stream>>>(w_gcn_kmer, wt_gcn_kmer, 340, 340, KMER_P, KMER_P);
  k_tcast_pad<<<nb((size_t)DOC_P*DOC_P,  T), T, 0, stream>>>(w_gcn_doc,  wt_gcn_doc,  256, 256, DOC_P,  DOC_P);
  k_tcast_pad<<<nb((size_t)ROLE_P*ROLE_P,T), T, 0, stream>>>(w_gcn_role, wt_gcn_role, 128, 128, ROLE_P, ROLE_P);
  k_tcast_pad<<<nb((size_t)256*CTD_P,  T), T, 0, stream>>>(w_p_ctd,  wt_p_ctd,  300, 256, CTD_P,  256);
  k_tcast_pad<<<nb((size_t)256*KMER_P, T), T, 0, stream>>>(w_p_kmer, wt_p_kmer, 340, 256, KMER_P, 256);
  k_tcast_pad<<<nb((size_t)256*DOC_P,  T), T, 0, stream>>>(w_p_doc,  wt_p_doc,  256, 256, DOC_P,  256);
  k_tcast_pad<<<nb((size_t)256*ROLE_P, T), T, 0, stream>>>(w_p_role, wt_p_role, 128, 256, ROLE_P, 256);
  k_tcast_pad<<<nb((size_t)256*CTD_P,  T), T, 0, stream>>>(w_pj_ctd,  wt_pj_ctd,  300, 256, CTD_P,  256);
  k_tcast_pad<<<nb((size_t)256*DOC_P,  T), T, 0, stream>>>(w_pj_doc,  wt_pj_doc,  256, 256, DOC_P,  256);
  k_tcast_pad<<<nb((size_t)256*KMER_P, T), T, 0, stream>>>(w_pj_kmer, wt_pj_kmer, 340, 256, KMER_P, 256);
  k_tcast_pad<<<nb((size_t)256*ROLE_P, T), T, 0, stream>>>(w_pj_role, wt_pj_role, 128, 256, ROLE_P, 256);
  k_tcast_pad<<<nb((size_t)256*NN,     T), T, 0, stream>>>(w_pj_sim,  wt_pj_sim,  NN,  256, NN,     256);
  k_tcast_pad<<<nb((size_t)256*PJW,    T), T, 0, stream>>>(w_fc_ban,  wt_fc,      PJW, 256, PJW,    256);

  k_pad_bias<<<1, 512, 0, stream>>>(b_gcn_ctd,  bp_ctd,  300, CTD_P);
  k_pad_bias<<<1, 512, 0, stream>>>(b_gcn_kmer, bp_kmer, 340, KMER_P);

  k_fill1 <<<nb(NN, T), T, 0, stream>>>(deg, NN);
  k_degacc<<<nb(EE, T), T, 0, stream>>>(dstI, deg, EE);
  k_dinv  <<<nb(NN, T), T, 0, stream>>>(deg, NN);           // deg -> dinv

  // -------- phase B: 4 GCN branches -> x_m (d_out region 1) --------
  struct Br { const bf16* xb; const bf16* wt; const float* bg; int Fp;
              const bf16* wtp; const float* bp; int off; };
  const Br brs[4] = {
    { xb_ctd,  wt_gcn_ctd,  bp_ctd,     CTD_P,  wt_p_ctd,  b_p_ctd,  0   },
    { xb_kmer, wt_gcn_kmer, bp_kmer,    KMER_P, wt_p_kmer, b_p_kmer, 256 },
    { xb_doc,  wt_gcn_doc,  b_gcn_doc,  DOC_P,  wt_p_doc,  b_p_doc,  512 },
    { xb_role, wt_gcn_role, b_gcn_role, ROLE_P, wt_p_role, b_p_role, 768 },
  };
  for (int i = 0; i < 4; ++i) {
    const Br& b = brs[i];
    gemm(b.xb, b.wt, nullptr, hbuf, b.Fp, b.Fp, b.Fp, 0);                    // h = x @ w
    k_agg_init <<<nb((size_t)NN*b.Fp, T), T, 0, stream>>>(hbuf, deg, b.bg, agg, b.Fp);
    k_agg_edges<<<EE, 128, 0, stream>>>(srcI, dstI, hbuf, deg, agg, b.Fp);   // scatter
    k_cast_pad <<<nb((size_t)NN*b.Fp, T), T, 0, stream>>>(agg, aggb, NN, b.Fp, b.Fp, 1); // relu+bf16
    gemm(aggb, b.wtp, b.bp, out_xm, b.Fp, 256, 1024, b.off);                 // projection
  }

  // -------- phase C: raw projections -> pj [8000,1280] --------
  gemm(xb_ctd,  wt_pj_ctd,  b_pj_ctd,  pj, CTD_P,  256, PJW, 0);
  gemm(xb_doc,  wt_pj_doc,  b_pj_doc,  pj, DOC_P,  256, PJW, 256);
  gemm(xb_role, wt_pj_role, b_pj_role, pj, ROLE_P, 256, PJW, 512);
  gemm(xb_kmer, wt_pj_kmer, b_pj_kmer, pj, KMER_P, 256, PJW, 768);
  gemm(sim_bf,  wt_pj_sim,  b_pj_sim,  pj, NN,     256, PJW, 1024);  // dominant GEMM

  // XM_raw_p = pj @ w_fc_ban + b (d_out region 2)
  k_cast_pad<<<nb((size_t)NN*PJW, T), T, 0, stream>>>(pj, pj_bf, NN, PJW, PJW, 0);
  gemm(pj_bf, wt_fc, b_fc_ban, out_raw, PJW, 256, 256, 0);

  // -------- phase D: CBAM + 1x1 conv -> x.T (d_out region 0) --------
  k_ca_partial<<<dim3(64, 5), 256, 0, stream>>>(pj, psum, pmax);
  k_ca_finish <<<1, 32, 0, stream>>>(psum, pmax, ca_w1, ca_w2, ca, 64);
  k_spool<<<nb((size_t)FM*NN, T), T, 0, stream>>>(pj, ca, sbuf);
  k_sconv<<<nb((size_t)FM*NN, T), T, 0, stream>>>(sbuf, pj, ca, sa_w, sa_b,
                                                  w_cnn, b_cnn, out_xT);
}
